// RelationClassifier_51479478010294
// MI455X (gfx1250) — matrix operand order repaired and checked by reading, not compiled, not run on hardware
//
#include <hip/hip_runtime.h>
#include <hip/hip_bf16.h>

// Problem constants (from reference)
#define B_ 4
#define R_ 64
#define S_ 512
#define H_ 768
#define E_ 64
#define SEG_ 128
#define NT_ 6
#define NC_ 13
#define NP_ 3
#define REP_ 2560           // 3*H + 2*SEG
#define NPAD_ 32            // 6+13+3 = 22 cols padded to 32 for WMMA

typedef _Float16 v16h __attribute__((ext_vector_type(16)));
typedef _Float16 v8h  __attribute__((ext_vector_type(8)));
typedef float    v8f  __attribute__((ext_vector_type(8)));

// ---------------------------------------------------------------------------
// Kernel A: one block per (b, r). Masked channel-max over sequence + entity
// gathers; writes the 2560-wide feature row in f16 (WMMA input precision).
// Feature row layout: [0:768]=span[p0] [768:1536]=span[p1]
//                     [1536:1664]=seg[p0] [1664:1792]=seg[p1] [1792:2560]=ctx
// ---------------------------------------------------------------------------
__global__ __launch_bounds__(256) void ctx_gather_kernel(
    const float* __restrict__ hidden,      // [B,S,H]
    const int*   __restrict__ rel_pairs,   // [B,R,2]
    const int*   __restrict__ rel_masks,   // [B,R,S]
    const float* __restrict__ span,        // [B,E,H]
    const float* __restrict__ segs,        // [B,E,SEG]
    _Float16*    __restrict__ featH)       // [B*R, REP_]
{
    const int row = blockIdx.x;            // b*R + r
    const int b   = row / R_;
    const int tid = threadIdx.x;

    __shared__ int smask[S_];
    const int* mrow = rel_masks + (size_t)row * S_;
    smask[tid]       = mrow[tid];
    smask[tid + 256] = mrow[tid + 256];
    __syncthreads();

    // Per-thread channel-max over masked positions (3 channels per thread).
    float mx0 = -3.0e38f, mx1 = -3.0e38f, mx2 = -3.0e38f;
    int   has = 0;
    const float* hb = hidden + (size_t)b * S_ * H_;
    for (int s = 0; s < S_; ++s) {
        if (smask[s]) {                    // wave-uniform branch (SALU skip)
            has = 1;
            const float* hs = hb + (size_t)s * H_;
            mx0 = fmaxf(mx0, hs[tid]);
            mx1 = fmaxf(mx1, hs[tid + 256]);
            mx2 = fmaxf(mx2, hs[tid + 512]);
        }
    }

    _Float16* fr = featH + (size_t)row * REP_;
    fr[1792 + tid]       = (_Float16)(has ? mx0 : 0.0f);
    fr[1792 + tid + 256] = (_Float16)(has ? mx1 : 0.0f);
    fr[1792 + tid + 512] = (_Float16)(has ? mx2 : 0.0f);

    // Entity gathers
    const int p0 = rel_pairs[row * 2 + 0];
    const int p1 = rel_pairs[row * 2 + 1];
    const float* sp0 = span + ((size_t)b * E_ + p0) * H_;
    const float* sp1 = span + ((size_t)b * E_ + p1) * H_;
    for (int i = tid; i < 2 * H_; i += 256) {
        float v = (i < H_) ? sp0[i] : sp1[i - H_];
        fr[i] = (_Float16)v;
    }
    {
        const int i = tid;                 // 2*SEG == 256 exactly
        float v = (i < SEG_) ? segs[((size_t)b * E_ + p0) * SEG_ + i]
                             : segs[((size_t)b * E_ + p1) * SEG_ + (i - SEG_)];
        fr[1536 + i] = (_Float16)v;
    }
}

// ---------------------------------------------------------------------------
// Kernel P: pack the three head weights into one column-major f16 B panel
// Bcol[n][k], n in [0,32) (cols 22..31 zero), plus a 32-entry f32 bias.
// Column-major makes each lane's WMMA B fragment two contiguous 16B loads.
// ---------------------------------------------------------------------------
__global__ __launch_bounds__(256) void prep_weights_kernel(
    const float* __restrict__ Wt, const float* __restrict__ bt,
    const float* __restrict__ Wc, const float* __restrict__ bc,
    const float* __restrict__ Wp, const float* __restrict__ bp,
    _Float16* __restrict__ Bcol,          // [32][REP_]
    float*    __restrict__ bias32)        // [32]
{
    const int idx = blockIdx.x * blockDim.x + threadIdx.x;
    if (idx < NPAD_ * REP_) {
        const int n = idx / REP_;
        const int k = idx % REP_;
        float v = 0.0f;
        if      (n < NT_)           v = Wt[k * NT_ + n];
        else if (n < NT_ + NC_)     v = Wc[k * NC_ + (n - NT_)];
        else if (n < NT_ + NC_ + NP_) v = Wp[k * NP_ + (n - NT_ - NC_)];
        Bcol[(size_t)n * REP_ + k] = (_Float16)v;
    }
    if (idx < NPAD_) {
        float v = 0.0f;
        if      (idx < NT_)             v = bt[idx];
        else if (idx < NT_ + NC_)       v = bc[idx - NT_];
        else if (idx < NT_ + NC_ + NP_) v = bp[idx - NT_ - NC_];
        bias32[idx] = v;
    }
}

// ---------------------------------------------------------------------------
// Kernel G: fused 3-head GEMM. 16 blocks (M tiles of 16 rows) x 2 waves
// (N tiles of 16 cols). Each wave runs an 80-step K loop of
// v_wmma_f32_16x16x32_f16, then bias-adds and scatters D into the three
// concatenated output segments.
//
// ISA fragment layouts (wave32):
//   A 16x32 f16 : lane L holds row M=L%16; halves e<8 -> K=(e&7)+8*(L>=16),
//                 e>=8 -> +16. => two contiguous v8h loads at +lhi*8, +16.
//   B 32x16 f16 : lane L holds col N=L%16; same K order as A => two
//                 contiguous v8h loads at +lhi*8, +16.
//   C/D 16x16 f32: VGPR i -> M = i + 8*(L>=16), N = L%16.
// ---------------------------------------------------------------------------
__global__ __launch_bounds__(64) void wmma_heads_kernel(
    const _Float16* __restrict__ featH,   // [256][REP_]
    const _Float16* __restrict__ Bcol,    // [32][REP_]
    const float*    __restrict__ bias32,  // [32]
    float*          __restrict__ out)     // [1536 | 3328 | 768] concat
{
    const int mtile = blockIdx.x;         // 0..15
    const int tid   = threadIdx.x;
    const int w     = tid >> 5;           // wave -> N tile (0 or 1)
    const int lane  = tid & 31;
    const int lhi   = (lane >> 4) & 1;
    const int l15   = lane & 15;

    const _Float16* arow = featH + (size_t)(mtile * 16 + l15) * REP_ + lhi * 8;
    const int ncol = w * 16 + l15;
    const _Float16* bptr = Bcol + (size_t)ncol * REP_ + lhi * 8;

    v8f acc = {};
    for (int kk = 0; kk < REP_; kk += 32) {
        v8h alo = *(const v8h*)(arow + kk);
        v8h ahi = *(const v8h*)(arow + kk + 16);
        v16h a  = __builtin_shufflevector(alo, ahi,
                      0, 1, 2, 3, 4, 5, 6, 7, 8, 9, 10, 11, 12, 13, 14, 15);
        v8h blo = *(const v8h*)(bptr + kk);
        v8h bhi = *(const v8h*)(bptr + kk + 16);
        v16h bfrag = __builtin_shufflevector(blo, bhi,
                      0, 1, 2, 3, 4, 5, 6, 7, 8, 9, 10, 11, 12, 13, 14, 15);
        acc = __builtin_amdgcn_wmma_f32_16x16x32_f16(
                  false, a, false, bfrag, (short)0, acc, false, false);
    }

    // Bias + scatter to the three concatenated logit outputs.
    const float bias = bias32[ncol];
    #pragma unroll
    for (int i = 0; i < 8; ++i) {
        const int mrow = mtile * 16 + i + lhi * 8;      // global feature row
        const float v  = acc[i] + bias;
        if (ncol < NT_) {
            out[mrow * NT_ + ncol] = v;                                   // type
        } else if (ncol < NT_ + NC_) {
            out[B_ * R_ * NT_ + mrow * NC_ + (ncol - NT_)] = v;           // cate
        } else if (ncol < NT_ + NC_ + NP_) {
            out[B_ * R_ * (NT_ + NC_) + mrow * NP_ + (ncol - NT_ - NC_)] = v; // pola
        }
        // ncol >= 22: padded column, discard
    }
}

// ---------------------------------------------------------------------------
extern "C" void kernel_launch(void* const* d_in, const int* in_sizes, int n_in,
                              void* d_out, int out_size, void* d_ws, size_t ws_size,
                              hipStream_t stream) {
    const float* hidden    = (const float*)d_in[0];
    const int*   rel_pairs = (const int*)  d_in[1];
    const int*   rel_masks = (const int*)  d_in[2];
    // d_in[3] = rel_sizes : unused by reference
    const float* span      = (const float*)d_in[4];
    const float* segs      = (const float*)d_in[5];
    const float* Wt        = (const float*)d_in[6];
    const float* bt        = (const float*)d_in[7];
    const float* Wc        = (const float*)d_in[8];
    const float* bc        = (const float*)d_in[9];
    const float* Wp        = (const float*)d_in[10];
    const float* bp        = (const float*)d_in[11];

    char* ws = (char*)d_ws;
    _Float16* featH  = (_Float16*)ws;                                 // 256*2560*2 = 1310720 B
    _Float16* Bcol   = (_Float16*)(ws + 1310720);                     // 32*2560*2  = 163840 B
    float*    bias32 = (float*)   (ws + 1310720 + 163840);            // 128 B
    float*    out    = (float*)d_out;

    // Feature build (ctx max + gathers) and weight packing are independent.
    ctx_gather_kernel<<<B_ * R_, 256, 0, stream>>>(hidden, rel_pairs, rel_masks,
                                                   span, segs, featH);
    prep_weights_kernel<<<(NPAD_ * REP_ + 255) / 256, 256, 0, stream>>>(
        Wt, bt, Wc, bc, Wp, bp, Bcol, bias32);

    // Fused 3-head GEMM via v_wmma_f32_16x16x32_f16.
    wmma_heads_kernel<<<16, 64, 0, stream>>>(featH, Bcol, bias32, out);
}
